// VQStepWiseTransformer_9440338116865
// MI455X (gfx1250) — matrix operand order repaired and checked by reading, not compiled, not run on hardware
//
#include <hip/hip_runtime.h>
#include <hip/hip_fp16.h>
#include <cmath>

// ---------------- model constants ----------------
#define B_      256
#define T_      24
#define TRANS_  40
#define OBS_    30
#define NE_     512
#define NH_     8
#define HD_     64
#define KCB_    512
#define LAT_    512
#define LSTEP_  3
#define MROWS_  (B_ * T_)          // 6144
#define SROWS_  (B_ * (T_/LSTEP_)) // 2048
#define NPAD_   64                 // pred head padded 40 -> 64

typedef _Float16 v16h  __attribute__((ext_vector_type(16)));
typedef __fp16   v2fp16 __attribute__((ext_vector_type(2)));   // cvt_pkrtz result type
typedef float    v8f   __attribute__((ext_vector_type(8)));

__device__ __forceinline__ float gelu_tanh(float x) {
    float x3 = x * x * x;
    return 0.5f * x * (1.0f + tanhf(0.7978845608028654f * (x + 0.044715f * x3)));
}

// single-instruction packed f32->f16 conversion (v_cvt_pk_rtz_f16_f32)
__device__ __forceinline__ unsigned long long pack4h(float4 v) {
    union { v2fp16 h2[2]; unsigned long long q; } t;
    t.h2[0] = __builtin_amdgcn_cvt_pkrtz(v.x, v.y);
    t.h2[1] = __builtin_amdgcn_cvt_pkrtz(v.z, v.w);
    return t.q;
}

// =====================================================================
// Exact-tiling WMMA GEMM:  C = epilogue(A[MxK] @ W + bias) (+R)
//   Requires M%128==0, N%64==0, K%32==0 (all call sites satisfy this;
//   the N=40 pred head is zero-padded to 64 beforehand).
//   TRANSB==0: W is [K,N] row-major;  TRANSB==1: W is [N,K] (use W^T).
// Block tile 128x64, BK=32, 8 wave32 waves (4x2), 2x2 16x16 WMMA tiles/wave.
// f32 in HBM; v_cvt_pk_rtz f32->f16 into LDS; f32 accumulate.
// Software-pipelined: next tile's global_load_b128 issued before the WMMAs.
// =====================================================================
template<int TRANSB>
__global__ __launch_bounds__(256) void k_gemm(
    const float* __restrict__ A, const float* __restrict__ W,
    const float* __restrict__ bias, const float* __restrict__ R,
    float* __restrict__ C, int N, int K, int act)
{
    constexpr int BM = 128, BN = 64, BK = 32;
    constexpr int LDA = 36, LDB = 36;   // halves; 72B rows (8B aligned), 18-dword
                                        // stride -> conflict-free frag reads
    __shared__ __align__(16) _Float16 As[BM * LDA];   // [m][k]
    __shared__ __align__(16) _Float16 Bs[BN * LDB];   // [n][k]

    const int tid  = threadIdx.x;
    const int wave = tid >> 5, lane = tid & 31;
    const int wm = wave & 3, wn = wave >> 2;          // 4x2 wave grid
    const int l16 = lane & 15, grp = lane >> 4;
    const int m0 = blockIdx.x * BM, n0 = blockIdx.y * BN;

    // staging coordinates (float4 granularity)
    const int ar0 = tid >> 3, ak4 = tid & 7;          // A: rows ar0+32e, k = ak4*4
    const int bk0 = tid >> 4, bn4 = tid & 15;         // B(noT): k = bk0+16e, n = bn4*4
    const int bn0 = tid >> 3, bk4 = tid & 7;          // B(T):   n = bn0+32e, k = bk4*4

    float4 ra[4], rb[2];

    auto loadG = [&](int k0) {
        #pragma unroll
        for (int e = 0; e < 4; e++)
            ra[e] = *(const float4*)&A[(size_t)(m0 + ar0 + 32 * e) * K + k0 + ak4 * 4];
        if (TRANSB == 0) {
            #pragma unroll
            for (int e = 0; e < 2; e++)
                rb[e] = *(const float4*)&W[(size_t)(k0 + bk0 + 16 * e) * N + n0 + bn4 * 4];
        } else {
            #pragma unroll
            for (int e = 0; e < 2; e++)
                rb[e] = *(const float4*)&W[(size_t)(n0 + bn0 + 32 * e) * K + k0 + bk4 * 4];
        }
    };
    auto storeLDS = [&]() {
        #pragma unroll
        for (int e = 0; e < 4; e++)
            *(unsigned long long*)&As[(ar0 + 32 * e) * LDA + ak4 * 4] = pack4h(ra[e]);
        if (TRANSB == 0) {
            #pragma unroll
            for (int e = 0; e < 2; e++) {
                v2fp16 lo = __builtin_amdgcn_cvt_pkrtz(rb[e].x, rb[e].y);
                v2fp16 hi = __builtin_amdgcn_cvt_pkrtz(rb[e].z, rb[e].w);
                Bs[(bn4 * 4 + 0) * LDB + bk0 + 16 * e] = (_Float16)lo[0];
                Bs[(bn4 * 4 + 1) * LDB + bk0 + 16 * e] = (_Float16)lo[1];
                Bs[(bn4 * 4 + 2) * LDB + bk0 + 16 * e] = (_Float16)hi[0];
                Bs[(bn4 * 4 + 3) * LDB + bk0 + 16 * e] = (_Float16)hi[1];
            }
        } else {
            #pragma unroll
            for (int e = 0; e < 2; e++)
                *(unsigned long long*)&Bs[(bn0 + 32 * e) * LDB + bk4 * 4] = pack4h(rb[e]);
        }
    };

    v8f c[2][2] = {};
    loadG(0);

    for (int k0 = 0; k0 < K; k0 += BK) {
        storeLDS();
        __syncthreads();

        const bool more = (k0 + BK) < K;
        if (more) {
            loadG(k0 + BK);                           // pipeline: 1 tile in regs
            if (k0 + 2 * BK < K) {                    // L2 prefetch 2 tiles ahead
                if (TRANSB == 0)
                    __builtin_prefetch(&W[(size_t)(k0 + 2 * BK + bk0) * N + n0 + bn4 * 4], 0, 1);
                else
                    __builtin_prefetch(&W[(size_t)(n0 + bn0) * K + k0 + 2 * BK + bk4 * 4], 0, 1);
                __builtin_prefetch(&A[(size_t)(m0 + ar0) * K + k0 + 2 * BK + ak4 * 4], 0, 1);
            }
        }

        // ---- fragments (ISA 7.12.2 16-bit layouts) ----
        union Fr { v16h v; unsigned u[8]; };
        Fr a[2], b[2];
        #pragma unroll
        for (int i = 0; i < 2; i++) {                 // A: row=l16, K interleaved per lane-group
            const unsigned* p = (const unsigned*)(As + (wm * 32 + i * 16 + l16) * LDA);
            #pragma unroll
            for (int r = 0; r < 8; r++) {
                int kk = grp * 8 + 2 * r + ((r >= 4) ? 8 : 0);
                a[i].u[r] = p[kk >> 1];
            }
        }
        #pragma unroll
        for (int j = 0; j < 2; j++) {                 // B: col=l16, K interleaved per lane-group (same as A)
            const unsigned* p = (const unsigned*)(Bs + (wn * 32 + j * 16 + l16) * LDB);
            #pragma unroll
            for (int r = 0; r < 8; r++) b[j].u[r] = p[grp * 4 + r + ((r >= 4) ? 4 : 0)];
        }
        #pragma unroll
        for (int i = 0; i < 2; i++)
            #pragma unroll
            for (int j = 0; j < 2; j++)
                c[i][j] = __builtin_amdgcn_wmma_f32_16x16x32_f16(
                    false, a[i].v, false, b[j].v, (short)0, c[i][j], false, false);
        __syncthreads();
    }

    // ---- epilogue: C/D layout -> row = r + 8*grp, col = l16 ----
    #pragma unroll
    for (int i = 0; i < 2; i++) {
        #pragma unroll
        for (int j = 0; j < 2; j++) {
            int nn = n0 + wn * 32 + j * 16 + l16;
            float bv = bias ? bias[nn] : 0.f;
            #pragma unroll
            for (int r = 0; r < 8; r++) {
                int mm = m0 + wm * 32 + i * 16 + grp * 8 + r;
                float v = c[i][j][r] + bv;
                if (act) v = gelu_tanh(v);
                if (R)   v += R[(size_t)mm * N + nn];
                C[(size_t)mm * N + nn] = v;
            }
        }
    }
}

// =====================================================================
// LayerNorm: one wave32 per 512-wide row
// =====================================================================
__global__ __launch_bounds__(256) void k_ln(const float* __restrict__ X,
                                            const float* __restrict__ w,
                                            const float* __restrict__ b,
                                            float* __restrict__ O, int rows)
{
    int row = blockIdx.x * 8 + (threadIdx.x >> 5);
    int lane = threadIdx.x & 31;
    if (row >= rows) return;
    const float* x = X + (size_t)row * NE_;
    float v[16], s = 0.f;
    #pragma unroll
    for (int i = 0; i < 16; i++) { v[i] = x[lane * 16 + i]; s += v[i]; }
    #pragma unroll
    for (int m = 16; m >= 1; m >>= 1) s += __shfl_xor(s, m);
    float mean = s * (1.f / NE_);
    float q = 0.f;
    #pragma unroll
    for (int i = 0; i < 16; i++) { float d = v[i] - mean; q += d * d; }
    #pragma unroll
    for (int m = 16; m >= 1; m >>= 1) q += __shfl_xor(q, m);
    float rstd = rsqrtf(q * (1.f / NE_) + 1e-5f);
    #pragma unroll
    for (int i = 0; i < 16; i++) {
        int n = lane * 16 + i;
        O[(size_t)row * NE_ + n] = (v[i] - mean) * rstd * w[n] + b[n];
    }
}

// =====================================================================
// Causal attention, T=24, HD=64: one wave32 per (batch, head)
// =====================================================================
__global__ __launch_bounds__(32) void k_attn(const float* __restrict__ Q,
                                             const float* __restrict__ Kx,
                                             const float* __restrict__ V,
                                             float* __restrict__ Y)
{
    int bh = blockIdx.x;
    int b = bh >> 3, h = bh & 7;
    __shared__ float Qs[T_][HD_ + 1];   // +1 pad: distinct banks across q-rows
    __shared__ float Ks[T_][HD_];       // broadcast reads
    __shared__ float Vs[T_][HD_];
    int lane = threadIdx.x;
    const size_t base = (size_t)b * T_ * NE_ + (size_t)h * HD_;
    for (int i = lane; i < T_ * HD_; i += 32) {
        int t = i >> 6, d = i & 63;
        size_t g = base + (size_t)t * NE_ + d;
        Qs[t][d] = Q[g]; Ks[t][d] = Kx[g]; Vs[t][d] = V[g];
    }
    __syncthreads();
    if (lane < T_) {
        int t = lane;
        float s[T_], mx = -1e30f;
        #pragma unroll
        for (int k = 0; k < T_; k++) {
            float acc = 0.f;
            #pragma unroll
            for (int d = 0; d < HD_; d++) acc += Qs[t][d] * Ks[k][d];
            acc *= 0.125f;                      // 1/sqrt(64)
            if (k > t) acc = -1e9f;             // causal mask (matches reference)
            s[k] = acc; mx = fmaxf(mx, acc);
        }
        float sum = 0.f;
        #pragma unroll
        for (int k = 0; k < T_; k++) { float e = __expf(s[k] - mx); s[k] = e; sum += e; }
        float inv = 1.f / sum;
        #pragma unroll
        for (int d = 0; d < HD_; d++) {
            float acc = 0.f;
            #pragma unroll
            for (int k = 0; k < T_; k++) acc += s[k] * Vs[k][d];
            Y[base + (size_t)t * NE_ + d] = acc * inv;
        }
    }
}

// =====================================================================
// Small fused kernels
// =====================================================================
// x[b,t,:] = (mji[b,t,:] @ embed_w) + embed_b + pos[t,:]   (channel 0 masked)
__global__ __launch_bounds__(256) void k_embed(const float* __restrict__ JI,
                                               const float* __restrict__ ew,
                                               const float* __restrict__ eb,
                                               const float* __restrict__ pos,
                                               float* __restrict__ X)
{
    int row = blockIdx.x;                 // b*24 + t
    int t = row % T_;
    __shared__ float in[TRANS_];
    if (threadIdx.x < TRANS_) {
        float v = JI[(size_t)row * TRANS_ + threadIdx.x];
        in[threadIdx.x] = (threadIdx.x == 0) ? 0.f : v;   // mask[:, :, 0] = 0
    }
    __syncthreads();
    for (int n = threadIdx.x; n < NE_; n += 256) {
        float acc = eb[n] + pos[(size_t)t * NE_ + n];
        #pragma unroll
        for (int c = 1; c < TRANS_; c++) acc += in[c] * ew[(size_t)c * NE_ + n];
        X[(size_t)row * NE_ + n] = acc;
    }
}

// MaxPool1d over latent_step=3
__global__ void k_pool(const float* __restrict__ X, float* __restrict__ P, int n)
{
    for (int i = blockIdx.x * blockDim.x + threadIdx.x; i < n; i += gridDim.x * blockDim.x) {
        int m = i >> 9, c = i & 511;       // NE=512
        int b = m >> 3, s = m & 7;
        size_t base = ((size_t)(b * T_ + s * LSTEP_)) * NE_ + c;
        P[i] = fmaxf(X[base], fmaxf(X[base + NE_], X[base + 2 * NE_]));
    }
}

// codebook row squared norms
__global__ __launch_bounds__(256) void k_cbnorm(const float* __restrict__ cb,
                                                float* __restrict__ cbn)
{
    int row = blockIdx.x * 8 + (threadIdx.x >> 5);
    int lane = threadIdx.x & 31;
    if (row >= KCB_) return;
    const float* c = cb + (size_t)row * LAT_;
    float s = 0.f;
    #pragma unroll
    for (int i = 0; i < 16; i++) { float v = c[lane + i * 32]; s += v * v; }
    #pragma unroll
    for (int m = 16; m >= 1; m >>= 1) s += __shfl_xor(s, m);
    if (lane == 0) cbn[row] = s;
}

// argmin(||c||^2 - 2 x.c) per row + gather codebook row (one wave per row)
__global__ __launch_bounds__(256) void k_vq(const float* __restrict__ S,
                                            const float* __restrict__ cbn,
                                            const float* __restrict__ cb,
                                            float* __restrict__ zq, int rows)
{
    int row = blockIdx.x * 8 + (threadIdx.x >> 5);
    int lane = threadIdx.x & 31;
    if (row >= rows) return;
    float best = 3.402823e38f; int bi = 0x7fffffff;
    #pragma unroll
    for (int i = 0; i < 16; i++) {
        int j = lane + i * 32;
        float d = cbn[j] - 2.f * S[(size_t)row * KCB_ + j];
        if (d < best || (d == best && j < bi)) { best = d; bi = j; }
    }
    #pragma unroll
    for (int m = 16; m >= 1; m >>= 1) {
        float ov = __shfl_xor(best, m);
        int   oi = __shfl_xor(bi, m);
        if (ov < best || (ov == best && oi < bi)) { best = ov; bi = oi; }
    }
    const float* c = cb + (size_t)bi * LAT_;
    float* z = zq + (size_t)row * LAT_;
    #pragma unroll
    for (int i = 0; i < 16; i++) z[lane + i * 32] = c[lane + i * 32];
}

// add the K=30 state-conditioning part of the mix GEMM (in-place)
__global__ __launch_bounds__(256) void k_mixstate(const float* __restrict__ state,
                                                  const float* __restrict__ mixw,
                                                  float* __restrict__ MIX)
{
    int row = blockIdx.x;                 // b*8 + s
    int b = row >> 3;
    __shared__ float st[OBS_];
    if (threadIdx.x < OBS_) st[threadIdx.x] = state[(size_t)b * OBS_ + threadIdx.x];
    __syncthreads();
    for (int n = threadIdx.x; n < NE_; n += 256) {
        float acc = MIX[(size_t)row * NE_ + n];
        #pragma unroll
        for (int c = 0; c < OBS_; c++) acc += st[c] * mixw[(size_t)c * NE_ + n];
        MIX[(size_t)row * NE_ + n] = acc;
    }
}

// decoder input: repeat_interleave(3) + positional embedding
__global__ void k_build(const float* __restrict__ MIX, const float* __restrict__ pos,
                        float* __restrict__ X, int n)
{
    for (int i = blockIdx.x * blockDim.x + threadIdx.x; i < n; i += gridDim.x * blockDim.x) {
        int m = i >> 9, c = i & 511;
        int b = m / T_, t = m % T_, s = t / LSTEP_;
        X[i] = MIX[((size_t)(b * 8 + s)) * NE_ + c] + pos[(size_t)t * NE_ + c];
    }
}

// zero-pad pred head [512x40]->[512x64], [40]->[64] so the exact GEMM applies
__global__ void k_padw(const float* __restrict__ W, const float* __restrict__ b,
                       float* __restrict__ PW, float* __restrict__ PB)
{
    int i = blockIdx.x * blockDim.x + threadIdx.x;
    if (i < NE_ * NPAD_) {
        int k = i >> 6, n = i & 63;
        PW[i] = (n < TRANS_) ? W[(size_t)k * TRANS_ + n] : 0.f;
    }
    if (i < NPAD_) PB[i] = (i < TRANS_) ? b[i] : 0.f;
}

// final: read padded pred [6144x64], sigmoid last channel, add state to 1..30
__global__ void k_final(const float* __restrict__ P, const float* __restrict__ state,
                        float* __restrict__ out, int n)
{
    for (int i = blockIdx.x * blockDim.x + threadIdx.x; i < n; i += gridDim.x * blockDim.x) {
        int c = i % TRANS_;
        int bt = i / TRANS_;
        int b = bt / T_;
        float v = P[(size_t)bt * NPAD_ + c];
        if (c == TRANS_ - 1) v = 1.f / (1.f + __expf(-v));
        else if (c >= 1 && c <= OBS_) v += state[(size_t)b * OBS_ + (c - 1)];
        out[i] = v;
    }
}

// =====================================================================
// Host launch
// =====================================================================
extern "C" void kernel_launch(void* const* d_in, const int* in_sizes, int n_in,
                              void* d_out, int out_size, void* d_ws, size_t ws_size,
                              hipStream_t stream)
{
    auto IN = [&](int i) { return (const float*)d_in[i]; };
    // flatten order (dict insertion order, recursive):
    // 0 joined_inputs, 1 state, 2 embed_w, 3 embed_b, 4 pos,
    // 5..68 enc (4 blocks x 16), 69..132 dec, 133 cast_w, 134 cast_b,
    // 135 mix_w, 136 mix_b, 137 lnf_w, 138 lnf_b, 139 pred_w, 140 pred_b, 141 codebook
    const float* JI = IN(0);
    const float* ST = IN(1);
    const float* blkp[8][16];
    for (int l = 0; l < 4; l++)
        for (int j = 0; j < 16; j++) { blkp[l][j] = IN(5 + l * 16 + j); blkp[4 + l][j] = IN(69 + l * 16 + j); }

    // workspace layout (floats); MLP hidden + VQ-phase + pred-phase alias Q..Y
    const size_t U = (size_t)MROWS_ * NE_;        // 6144*512
    const size_t S3 = (size_t)SROWS_ * NE_;       // 2048*512 = U/3
    float* X   = (float*)d_ws;
    float* H   = X + U;
    float* Qb  = X + 2 * U;
    float* Kb  = X + 3 * U;
    float* Vb  = X + 4 * U;
    float* Yb  = X + 5 * U;
    float* MLP = Qb;                              // 6144*2048 spans Qb..Yb exactly
    float* POOL = Qb;                             // VQ-phase aliases (Q/K/V/Y dead then)
    float* MTF  = Qb + S3;
    float* Sc   = Qb + 2 * S3;
    float* ZQ   = Kb;
    float* MIX  = Kb + S3;
    float* CBN  = Kb + 2 * S3;
    float* PRED = Qb;                             // 6144*64 (pred-phase)
    float* PW   = Vb;                             // 512*64 padded pred_w
    float* PB   = Vb + NE_ * NPAD_;               // 64 padded pred_b

    auto gemm = [&](const float* A, const float* W, const float* bias, const float* R,
                    float* C, int M, int N, int K, int act) {
        dim3 g(M / 128, N / 64);
        k_gemm<0><<<g, dim3(256), 0, stream>>>(A, W, bias, R, C, N, K, act);
    };
    auto gemmT = [&](const float* A, const float* W, float* C, int M, int N, int K) {
        dim3 g(M / 128, N / 64);
        k_gemm<1><<<g, dim3(256), 0, stream>>>(A, W, nullptr, nullptr, C, N, K, 0);
    };

    auto run_block = [&](const float* const* p) {
        k_ln<<<MROWS_ / 8, 256, 0, stream>>>(X, p[0], p[1], H, MROWS_);
        gemm(H, p[2], p[3], nullptr, Qb, MROWS_, NE_, NE_, 0);          // Q
        gemm(H, p[4], p[5], nullptr, Kb, MROWS_, NE_, NE_, 0);          // K
        gemm(H, p[6], p[7], nullptr, Vb, MROWS_, NE_, NE_, 0);          // V
        k_attn<<<B_ * NH_, 32, 0, stream>>>(Qb, Kb, Vb, Yb);
        gemm(Yb, p[8], p[9], X, X, MROWS_, NE_, NE_, 0);                // x += y@wo + bo
        k_ln<<<MROWS_ / 8, 256, 0, stream>>>(X, p[10], p[11], H, MROWS_);
        gemm(H, p[12], p[13], nullptr, MLP, MROWS_, 4 * NE_, NE_, 1);   // gelu(h@w1+b1)
        gemm(MLP, p[14], p[15], X, X, MROWS_, NE_, 4 * NE_, 0);         // x += m@w2 + b2
    };

    // ---- encoder (dead un-masked encode elided: EMA update is a no-op in eval) ----
    k_embed<<<MROWS_, 256, 0, stream>>>(JI, IN(2), IN(3), IN(4), X);
    for (int l = 0; l < 4; l++) run_block(blkp[l]);

    // ---- pool + cast + VQ ----
    k_pool<<<1024, 256, 0, stream>>>(X, POOL, SROWS_ * NE_);
    gemm(POOL, IN(133), IN(134), nullptr, MTF, SROWS_, LAT_, NE_, 0);
    k_cbnorm<<<KCB_ / 8, 256, 0, stream>>>(IN(141), CBN);
    gemmT(MTF, IN(141), Sc, SROWS_, KCB_, LAT_);                        // mtf @ cb^T
    k_vq<<<SROWS_ / 8, 256, 0, stream>>>(Sc, CBN, IN(141), ZQ, SROWS_);

    // ---- decoder ----
    gemm(ZQ, IN(135) + (size_t)OBS_ * NE_, IN(136), nullptr, MIX, SROWS_, NE_, LAT_, 0);
    k_mixstate<<<SROWS_, 256, 0, stream>>>(ST, IN(135), MIX);
    k_build<<<1536, 256, 0, stream>>>(MIX, IN(4), X, MROWS_ * NE_);
    for (int l = 0; l < 4; l++) run_block(blkp[4 + l]);

    // ---- head: pad pred weights to N=64, exact GEMM, then fix-up ----
    k_ln<<<MROWS_ / 8, 256, 0, stream>>>(X, IN(137), IN(138), H, MROWS_);
    k_padw<<<(NE_ * NPAD_) / 256, 256, 0, stream>>>(IN(139), IN(140), PW, PB);
    gemm(H, PW, PB, nullptr, PRED, MROWS_, NPAD_, NE_, 0);
    k_final<<<480, 256, 0, stream>>>(PRED, ST, (float*)d_out, B_ * T_ * TRANS_);

    (void)in_sizes; (void)n_in; (void)out_size; (void)ws_size;
}
